// NccCostVolume_52690658787613
// MI455X (gfx1250) — matrix operand order repaired and checked by reading, not compiled, not run on hardware
//
#include <hip/hip_runtime.h>
#include <math.h>

// Problem constants (match reference)
#define B_   2
#define C_   16
#define H_   128
#define W_   256
#define CW_  32
#define HR_  3
#define VR_  3
#define P_   (C_ * (2*VR_+1) * (2*HR_+1))   // 784
#define WE_  (W_ + CW_ - 1)                 // 287 (left extended width)
#define EPS_ 1e-9f

typedef __attribute__((ext_vector_type(16))) _Float16 v16h;
typedef __attribute__((ext_vector_type(8)))  float    v8f;

// ---- workspace layout (float offsets) ----
#define OFF_SP   0                              // vertically-summed correlation S'[b,h,w,32]
#define N_SP     (B_*H_*W_*CW_)
#define OFF_CSL  (OFF_SP  + N_SP)               // per-pixel channel sum, left
#define OFF_CQL  (OFF_CSL + B_*H_*W_)           // per-pixel channel sumsq, left
#define OFF_CSR  (OFF_CQL + B_*H_*W_)
#define OFF_CQR  (OFF_CSR + B_*H_*W_)
#define OFF_PSL  (OFF_CQR + B_*H_*W_)           // patch sum, left (width WE_)
#define OFF_INL  (OFF_PSL + B_*H_*WE_)          // 1/patch-norm, left
#define OFF_PSR  (OFF_INL + B_*H_*WE_)
#define OFF_INR  (OFF_PSR + B_*H_*W_)

__device__ __forceinline__ int clampi(int v, int lo, int hi) {
  return v < lo ? lo : (v > hi ? hi : v);
}

// =====================================================================
// Kernel 1: banded correlation via WMMA.
// One wave per (b, y, 16-wide x-tile). K folds (dy in [-3,3]) x (c in
// [0,16)) -> K=112, padded to 128 = 4 chained v_wmma_f32_16x16x32_f16.
// Three B-tiles (u-tiles) cover the 32-disparity band.
// A layout: c = t, dy-slot = 2q + hi (same as B)
// B layout: c = t, dy-slot = 2q + hi.
// Slot 7 of the row tables is the always-zero K-padding slot.
// All loads are unconditional global_load (clamped row/col + float mask),
// addressed as 32-bit element offsets off the kernel-arg pointers so the
// compiler keeps the global address space (no flat fallback).
// =====================================================================
__global__ __launch_bounds__(256)
void ncc_wmma_vertcc(const float* __restrict__ L, const float* __restrict__ R,
                     float* __restrict__ Sp) {
  const int wave = blockIdx.x * (blockDim.x >> 5) + (threadIdx.x >> 5);
  const int lane = threadIdx.x & 31;
  const int hi   = lane >> 4;
  const int m    = lane & 15;

  const int b   = wave / (H_ * (W_/16));
  const int rem = wave % (H_ * (W_/16));
  const int y   = rem / (W_/16);
  const int x0  = (rem % (W_/16)) * 16;

  const int imgBase = b * C_ * H_ * W_;     // batch offset (elements)

  // clamped per-dy row offsets + validity masks (slot 7 == zero pad slot)
  int rowOff[8]; float mrow[8];
#pragma unroll
  for (int dy = 0; dy < 8; ++dy) {
    const int  yy = y + dy - VR_;
    const bool ok = (dy < 7) && (yy >= 0) && (yy < H_);
    const int yyc = clampi(yy, 0, H_ - 1);
    mrow[dy]   = ok ? 1.0f : 0.0f;
    rowOff[dy] = imgBase + yyc * W_;
  }

  // per-lane constant offsets
  const int offA = (x0 + m);                     // A: column
  int ucol[3]; float cmask[3];
#pragma unroll
  for (int tt = 0; tt < 3; ++tt) {
    const int u = x0 + 16*tt + m;                // left column (extended)
    cmask[tt] = (u < W_) ? 1.0f : 0.0f;
    ucol[tt]  = (u < W_) ? u : (W_ - 1);
  }

  v8f acc0 = {}; v8f acc1 = {}; v8f acc2 = {};

#pragma unroll
  for (int q = 0; q < 4; ++q) {
    // ---- A operand (right image): 16 unconditional loads, imm offsets
    const int aLo = rowOff[2*q + hi] + offA;     // dy-slot 2q+hi, channels 0-15
    float fa[16];
#pragma unroll
    for (int t = 0; t < 16; ++t) fa[t]    = R[aLo + t * (H_*W_)];
    const float maLo = mrow[2*q + hi], maHi = mrow[2*q + hi];
    v16h a;
#pragma unroll
    for (int t = 0; t < 8; ++t) {
      a[t]     = (_Float16)(fa[t]     * maLo);
      a[t + 8] = (_Float16)(fa[t + 8] * maHi);
    }

    // ---- B operands (left image), one per u-tile
    const int   bRow = hi ? rowOff[2*q + 1] : rowOff[2*q];
    const float mB0  = hi ? mrow[2*q + 1]   : mrow[2*q];
#pragma unroll
    for (int tt = 0; tt < 3; ++tt) {
      const int   pB = bRow + ucol[tt];
      const float mB = mB0 * cmask[tt];
      float fb[16];
#pragma unroll
      for (int t = 0; t < 16; ++t) fb[t] = L[pB + t * (H_*W_)];
      v16h bm;
#pragma unroll
      for (int t = 0; t < 16; ++t) bm[t] = (_Float16)(fb[t] * mB);

      if      (tt == 0) acc0 = __builtin_amdgcn_wmma_f32_16x16x32_f16(false, a, false, bm, (short)0, acc0, false, false);
      else if (tt == 1) acc1 = __builtin_amdgcn_wmma_f32_16x16x32_f16(false, a, false, bm, (short)0, acc1, false, false);
      else              acc2 = __builtin_amdgcn_wmma_f32_16x16x32_f16(false, a, false, bm, (short)0, acc2, false, false);
    }
  }

  // ---- branch-free band scatter.
  // tile1 always valid: d = 16+m-M in [1,31].
  // tile0 (m>=M) and tile2 (m<M) are complementary; d = (m-M)&31.
  const int rowIdx = ((b*H_ + y) * W_ + x0) * CW_;
#pragma unroll
  for (int r = 0; r < 8; ++r) {
    const int   M   = r + 8*hi;                  // right-x within tile
    const int   d02 = (m - M) & 31;
    const float v02 = (m >= M) ? acc0[r] : acc2[r];
    Sp[rowIdx + M * CW_ + d02]        = v02;
    Sp[rowIdx + M * CW_ + 16 + m - M] = acc1[r];
  }
}

// =====================================================================
// Kernel 2: per-pixel channel sum & sum-of-squares (over C=16)
// =====================================================================
__global__ void ncc_chansum(const float* __restrict__ img,
                            float* __restrict__ cs, float* __restrict__ cq) {
  const int i = blockIdx.x * blockDim.x + threadIdx.x;     // over B*H*W
  if (i >= B_*H_*W_) return;
  const int b  = i / (H_*W_);
  const int yx = i % (H_*W_);
  const int base = b * C_ * H_ * W_ + yx;
  float s = 0.0f, q = 0.0f;
#pragma unroll
  for (int c = 0; c < C_; ++c) { const float v = img[base + c*H_*W_]; s += v; q += v*v; }
  cs[i] = s; cq[i] = q;
}

// =====================================================================
// Kernel 3: 7x7 box of channel sums -> patch sum + 1/patch-norm.
// outW = WE_ for left (zero columns beyond W), W_ for right.
// =====================================================================
__global__ void ncc_patchstats(const float* __restrict__ cs, const float* __restrict__ cq,
                               float* __restrict__ ps, float* __restrict__ invn, int outW) {
  const int i = blockIdx.x * blockDim.x + threadIdx.x;     // over B*H*outW
  if (i >= B_*H_*outW) return;
  const int b   = i / (H_*outW);
  const int rem = i % (H_*outW);
  const int y   = rem / outW;
  const int x   = rem % outW;
  float s = 0.0f, q = 0.0f;
#pragma unroll
  for (int dy = -VR_; dy <= VR_; ++dy) {
    const int   yy  = y + dy;
    const float my  = (yy >= 0 && yy < H_) ? 1.0f : 0.0f;
    const int   yyc = clampi(yy, 0, H_ - 1);
#pragma unroll
    for (int dx = -HR_; dx <= HR_; ++dx) {
      const int   xx  = x + dx;
      const float w   = my * ((xx >= 0 && xx < W_) ? 1.0f : 0.0f);
      const int   xxc = clampi(xx, 0, W_ - 1);
      const int   j   = (b*H_ + yyc)*W_ + xxc;
      s += cs[j] * w; q += cq[j] * w;
    }
  }
  ps[i] = s;
  const float var = q - s * s * (1.0f / (float)P_);
  invn[i] = rsqrtf(var + EPS_);
}

// =====================================================================
// Kernel 4: horizontal 7-tap box of S' + analytic NCC normalization.
// cost = (S - psL*psR/P) * invnL * invnR ; output layout [b,h,w,CW]
// =====================================================================
__global__ void ncc_finalize(const float* __restrict__ Sp,
                             const float* __restrict__ psL, const float* __restrict__ inL,
                             const float* __restrict__ psR, const float* __restrict__ inR,
                             float* __restrict__ out) {
  const int i = blockIdx.x * blockDim.x + threadIdx.x;     // over B*H*W*CW
  if (i >= B_*H_*W_*CW_) return;
  const int d   = i & (CW_ - 1);
  const int pix = i >> 5;          // b*H*W + y*W + x
  const int x   = pix % W_;
  const int by  = pix / W_;        // b*H + y

  float S = 0.0f;
#pragma unroll
  for (int dx = -HR_; dx <= HR_; ++dx) {
    const int   xx  = x + dx;
    const float w   = (xx >= 0 && xx < W_) ? 1.0f : 0.0f;
    const int   xxc = clampi(xx, 0, W_ - 1);
    S += Sp[(by * W_ + xxc) * CW_ + d] * w;
  }
  const int xe = x + d;                       // left extended column (< WE_)
  const int li = by * WE_ + xe;
  const int ri = by * W_  + x;
  out[i] = (S - psL[li] * psR[ri] * (1.0f / (float)P_)) * inL[li] * inR[ri];
}

// =====================================================================
extern "C" void kernel_launch(void* const* d_in, const int* in_sizes, int n_in,
                              void* d_out, int out_size, void* d_ws, size_t ws_size,
                              hipStream_t stream) {
  (void)in_sizes; (void)n_in; (void)out_size; (void)ws_size;
  const float* L = (const float*)d_in[0];   // im_left  [B,C,H,W] f32
  const float* R = (const float*)d_in[1];   // im_right [B,C,H,W] f32
  float* out = (float*)d_out;               // [B,H,W,CW] f32
  float* ws  = (float*)d_ws;

  float* Sp  = ws + OFF_SP;
  float* csL = ws + OFF_CSL; float* cqL = ws + OFF_CQL;
  float* csR = ws + OFF_CSR; float* cqR = ws + OFF_CQR;
  float* psL = ws + OFF_PSL; float* inL = ws + OFF_INL;
  float* psR = ws + OFF_PSR; float* inR = ws + OFF_INR;

  // 1) WMMA banded correlation (4096 waves, 8 waves/block)
  const int nWaves = B_ * H_ * (W_/16);                 // 4096
  ncc_wmma_vertcc<<<nWaves/8, 256, 0, stream>>>(L, R, Sp);

  // 2) per-pixel channel sums
  const int nPix = B_ * H_ * W_;                        // 65536
  ncc_chansum<<<(nPix + 255)/256, 256, 0, stream>>>(L, csL, cqL);
  ncc_chansum<<<(nPix + 255)/256, 256, 0, stream>>>(R, csR, cqR);

  // 3) patch statistics (left over extended width, right over W)
  const int nL = B_ * H_ * WE_;                         // 73472
  ncc_patchstats<<<(nL   + 255)/256, 256, 0, stream>>>(csL, cqL, psL, inL, WE_);
  ncc_patchstats<<<(nPix + 255)/256, 256, 0, stream>>>(csR, cqR, psR, inR, W_);

  // 4) horizontal box + normalize -> output
  const int nOut = B_ * H_ * W_ * CW_;                  // 2097152
  ncc_finalize<<<(nOut + 255)/256, 256, 0, stream>>>(Sp, psL, inL, psR, inR, out);
}
